// FuseBaseNoSia_17239998726589
// MI455X (gfx1250) — matrix operand order repaired and checked by reading, not compiled, not run on hardware
//
#include <hip/hip_runtime.h>
#include <hip/hip_bf16.h>

// Problem constants (match reference)
#define NN      50000
#define EE      800000
#define DD      64
#define GG      512
#define NCLASS  10

typedef __attribute__((ext_vector_type(16))) _Float16 v16h;
typedef __attribute__((ext_vector_type(8)))  float    v8f;

// ---------------------------------------------------------------------------
// Degree count: one thread per edge, atomicAdd 1.0 to deg[dst[e]]
// ---------------------------------------------------------------------------
__global__ void deg_count_kernel(const int* __restrict__ dst, float* __restrict__ deg) {
    int e = blockIdx.x * blockDim.x + threadIdx.x;
    if (e >= EE) return;
    atomicAdd(&deg[dst[e]], 1.0f);
}

// deg[i] -> 1 / max(deg[i], 1)
__global__ void deg_invert_kernel(float* __restrict__ deg) {
    int i = blockIdx.x * blockDim.x + threadIdx.x;
    if (i >= NN) return;
    deg[i] = 1.0f / fmaxf(deg[i], 1.0f);
}

// ---------------------------------------------------------------------------
// Edge scatter: thread t handles edge e = t/16, quad q = t%16.
// agg[dst[e], 4q..4q+3] += x[src[e], 4q..4q+3]  (f32 atomics, L2-resident)
// ---------------------------------------------------------------------------
__global__ void scatter_kernel(const float* __restrict__ xsrc,
                               const int*   __restrict__ src,
                               const int*   __restrict__ dst,
                               float*       __restrict__ agg) {
    long long t = (long long)blockIdx.x * blockDim.x + threadIdx.x;
    int e = (int)(t >> 4);
    int q = (int)(t & 15);
    if (e >= EE) return;
    int s = src[e];
    int d = dst[e];
    const float4 v = *(const float4*)(xsrc + (size_t)s * DD + q * 4);
    float* ap = agg + (size_t)d * DD + q * 4;
    atomicAdd(ap + 0, v.x);
    atomicAdd(ap + 1, v.y);
    atomicAdd(ap + 2, v.z);
    atomicAdd(ap + 3, v.w);
}

// ---------------------------------------------------------------------------
// WMMA GEMM: xout = relu( (agg * inv_deg[row]) @ W + bias )
//   agg: [N,64] f32, W: [64,64] f32 row-major, bias: [64]
// One 256-thread block = 8 waves; each wave computes a 16x64 output tile via
// v_wmma_f32_16x16x32_f16 (2 K-steps x 4 N-tiles = 8 WMMAs), f32 accumulate.
// W is pre-packed into LDS in the exact WMMA B-fragment register layout so
// each lane fetches its 16 halves with one aligned 32-byte LDS vector load.
// ---------------------------------------------------------------------------
__global__ __launch_bounds__(256) void gemm_relu_wmma_kernel(
        const float* __restrict__ agg,
        const float* __restrict__ invdeg,
        const float* __restrict__ W,
        const float* __restrict__ bias,
        float*       __restrict__ xout) {
    // sWfrag[frag][lane] : frag = kc*4 + nt (kc in 0..1, nt in 0..3)
    __shared__ v16h  sWfrag[8][32];    // 8 KB, B fragments in register layout
    __shared__ float sBias[DD];

    // ---- Pack phase: 256 threads <-> 256 (fragment, lane) pairs.
    // Fragment (kc, nt), lane L, element j holds W[K][N] with
    //   K = kc*32 + 16*(L>>4) + j,  N = nt*16 + (L&15)
    {
        const int t    = threadIdx.x;
        const int frag = t >> 5;            // 0..7
        const int L    = t & 31;
        const int kc   = frag >> 2;         // 0..1
        const int nt   = frag & 3;          // 0..3
        const int kb   = kc * 32 + 16 * (L >> 4);
        const int n    = nt * 16 + (L & 15);
        v16h tmp;
        #pragma unroll
        for (int j = 0; j < 16; ++j) tmp[j] = (_Float16)W[(kb + j) * DD + n];
        sWfrag[frag][L] = tmp;
        if (t < DD) sBias[t] = bias[t];
    }
    __syncthreads();

    const int wave  = threadIdx.x >> 5;
    const int lane  = threadIdx.x & 31;
    const int mtile = blockIdx.x * 8 + wave;
    const int NTILES = NN / 16;                 // 3125 (N divisible by 16)
    if (mtile >= NTILES) return;

    const int m0    = mtile * 16;
    const int mlo   = lane & 15;
    const int khalf = lane >> 4;                // 0: lanes 0-15, 1: lanes 16-31
    const int mrow  = m0 + mlo;

    const float scale = invdeg[mrow];
    const float* arow = agg + (size_t)mrow * DD;

    v8f acc[4] = {};                            // 4 N-tiles of 16 cols each

    #pragma unroll
    for (int kc = 0; kc < 2; ++kc) {
        const int k0 = kc * 32;
        // ---- A fragment: 16x32 f16. Lane holds row mrow,
        //      K = k0+khalf*16+j (elems 0..15).
        v16h a;
        const int kg0 = k0 + khalf * 16;
        const int kg1 = kg0 + 8;
        float4 p0 = *(const float4*)(arow + kg0);
        float4 p1 = *(const float4*)(arow + kg0 + 4);
        float4 p2 = *(const float4*)(arow + kg1);
        float4 p3 = *(const float4*)(arow + kg1 + 4);
        a[0]  = (_Float16)(p0.x * scale); a[1]  = (_Float16)(p0.y * scale);
        a[2]  = (_Float16)(p0.z * scale); a[3]  = (_Float16)(p0.w * scale);
        a[4]  = (_Float16)(p1.x * scale); a[5]  = (_Float16)(p1.y * scale);
        a[6]  = (_Float16)(p1.z * scale); a[7]  = (_Float16)(p1.w * scale);
        a[8]  = (_Float16)(p2.x * scale); a[9]  = (_Float16)(p2.y * scale);
        a[10] = (_Float16)(p2.z * scale); a[11] = (_Float16)(p2.w * scale);
        a[12] = (_Float16)(p3.x * scale); a[13] = (_Float16)(p3.y * scale);
        a[14] = (_Float16)(p3.z * scale); a[15] = (_Float16)(p3.w * scale);

        #pragma unroll
        for (int nt = 0; nt < 4; ++nt) {
            // ---- B fragment: one aligned 32B vector load from packed LDS
            v16h b = sWfrag[kc * 4 + nt][lane];
            acc[nt] = __builtin_amdgcn_wmma_f32_16x16x32_f16(
                false, a, false, b, (short)0, acc[nt], false, false);
        }
    }

    // ---- Store with bias + ReLU. D layout: VGPR r -> M = m0 + r + 8*khalf,
    //      N = nt*16 + mlo.
    #pragma unroll
    for (int nt = 0; nt < 4; ++nt) {
        const int col = nt * 16 + mlo;
        const float bv = sBias[col];
        #pragma unroll
        for (int r = 0; r < 8; ++r) {
            const int m = m0 + r + 8 * khalf;
            float v = acc[nt][r] + bv;
            xout[(size_t)m * DD + col] = v > 0.0f ? v : 0.0f;
        }
    }
}

// ---------------------------------------------------------------------------
// Fuse branches + output projection + graph scatter:
//   h = (x0+x1)*0.5 ; out = h @ Wout + bout ; atomic into gsum/gcnt
// ---------------------------------------------------------------------------
__global__ void fuse_out_kernel(const float* __restrict__ x0,
                                const float* __restrict__ x1,
                                const float* __restrict__ Wout,   // [64,10]
                                const float* __restrict__ bout,   // [10]
                                const int*   __restrict__ bidx,
                                float* __restrict__ gsum,
                                float* __restrict__ gcnt) {
    int n = blockIdx.x * blockDim.x + threadIdx.x;
    if (n >= NN) return;
    float acc[NCLASS];
    #pragma unroll
    for (int c = 0; c < NCLASS; ++c) acc[c] = bout[c];
    const float* r0 = x0 + (size_t)n * DD;
    const float* r1 = x1 + (size_t)n * DD;
    for (int d = 0; d < DD; ++d) {
        float h = (r0[d] + r1[d]) * 0.5f;
        const float* wr = Wout + d * NCLASS;
        #pragma unroll
        for (int c = 0; c < NCLASS; ++c) acc[c] += h * wr[c];
    }
    int g = bidx[n];
    float* gp = gsum + (size_t)g * NCLASS;
    #pragma unroll
    for (int c = 0; c < NCLASS; ++c) atomicAdd(&gp[c], acc[c]);
    atomicAdd(&gcnt[g], 1.0f);
}

__global__ void finalize_kernel(const float* __restrict__ gsum,
                                const float* __restrict__ gcnt,
                                float* __restrict__ out) {
    int i = blockIdx.x * blockDim.x + threadIdx.x;
    if (i >= GG * NCLASS) return;
    int g = i / NCLASS;
    out[i] = gsum[i] / fmaxf(gcnt[g], 1.0f);
}

// ---------------------------------------------------------------------------
// Host launcher
// ---------------------------------------------------------------------------
extern "C" void kernel_launch(void* const* d_in, const int* in_sizes, int n_in,
                              void* d_out, int out_size, void* d_ws, size_t ws_size,
                              hipStream_t stream) {
    (void)in_sizes; (void)n_in; (void)out_size; (void)ws_size;

    const float* x     = (const float*)d_in[0];            // [N,64]
    const float* W0    = (const float*)d_in[1];            // [2,64,64]
    const float* b0    = (const float*)d_in[2];            // [2,64]
    const float* W1    = (const float*)d_in[3];            // [2,64,64]
    const float* b1    = (const float*)d_in[4];            // [2,64]
    const float* Wout  = (const float*)d_in[5];            // [64,10]
    const float* bout  = (const float*)d_in[6];            // [10]
    const int*   ei_sc = (const int*)d_in[7];              // [2,E]
    const int*   ei_fc = (const int*)d_in[8];              // [2,E]
    const int*   bidx  = (const int*)d_in[9];              // [N]
    float* out = (float*)d_out;                            // [G,10]

    // Workspace layout (floats)
    const size_t ND = (size_t)NN * DD;
    float* ws   = (float*)d_ws;
    float* x0b  = ws;                 // [N,64]
    float* x1b  = x0b + ND;           // [N,64]
    float* agg  = x1b + ND;           // [N,64]
    float* dsc  = agg + ND;           // [N]  degree -> inv_deg (sc)
    float* dfc  = dsc + NN;           // [N]  degree -> inv_deg (fc)
    float* gsum = dfc + NN;           // [G,10]
    float* gcnt = gsum + (size_t)GG * NCLASS;  // [G]

    const int TB = 256;
    const int degBlocks  = (EE + TB - 1) / TB;
    const int invBlocks  = (NN + TB - 1) / TB;
    const int scatBlocks = (int)(((long long)EE * 16 + TB - 1) / TB);
    const int gemmBlocks = (NN / 16 + 7) / 8;   // 8 waves/block, 16 rows/wave
    const int fuseBlocks = (NN + TB - 1) / TB;
    const int finBlocks  = (GG * NCLASS + TB - 1) / TB;

    // Zero degrees + graph accumulators (contiguous region)
    hipMemsetAsync(dsc, 0, (size_t)(2 * NN + GG * NCLASS + GG) * sizeof(float), stream);

    // Degrees per edge set, then invert (mean normalization)
    deg_count_kernel<<<degBlocks, TB, 0, stream>>>(ei_sc + EE, dsc);
    deg_count_kernel<<<degBlocks, TB, 0, stream>>>(ei_fc + EE, dfc);
    deg_invert_kernel<<<invBlocks, TB, 0, stream>>>(dsc);
    deg_invert_kernel<<<invBlocks, TB, 0, stream>>>(dfc);

    // ---- Branch 0 (structural edges), layers 0..1
    hipMemsetAsync(agg, 0, ND * sizeof(float), stream);
    scatter_kernel<<<scatBlocks, TB, 0, stream>>>(x, ei_sc, ei_sc + EE, agg);
    gemm_relu_wmma_kernel<<<gemmBlocks, TB, 0, stream>>>(agg, dsc, W0, b0, x0b);

    hipMemsetAsync(agg, 0, ND * sizeof(float), stream);
    scatter_kernel<<<scatBlocks, TB, 0, stream>>>(x0b, ei_sc, ei_sc + EE, agg);
    gemm_relu_wmma_kernel<<<gemmBlocks, TB, 0, stream>>>(agg, dsc, W0 + DD * DD, b0 + DD, x0b);

    // ---- Branch 1 (functional edges), layers 0..1
    hipMemsetAsync(agg, 0, ND * sizeof(float), stream);
    scatter_kernel<<<scatBlocks, TB, 0, stream>>>(x, ei_fc, ei_fc + EE, agg);
    gemm_relu_wmma_kernel<<<gemmBlocks, TB, 0, stream>>>(agg, dfc, W1, b1, x1b);

    hipMemsetAsync(agg, 0, ND * sizeof(float), stream);
    scatter_kernel<<<scatBlocks, TB, 0, stream>>>(x1b, ei_fc, ei_fc + EE, agg);
    gemm_relu_wmma_kernel<<<gemmBlocks, TB, 0, stream>>>(agg, dfc, W1 + DD * DD, b1 + DD, x1b);

    // ---- Fuse + projection + graph mean
    fuse_out_kernel<<<fuseBlocks, TB, 0, stream>>>(x0b, x1b, Wout, bout, bidx, gsum, gcnt);
    finalize_kernel<<<finBlocks, TB, 0, stream>>>(gsum, gcnt, out);
}
